// MemeSTN_17033840295983
// MI455X (gfx1250) — matrix operand order repaired and checked by reading, not compiled, not run on hardware
//
#include <hip/hip_runtime.h>
#include <hip/hip_bf16.h>
#include <math.h>

typedef __attribute__((ext_vector_type(16))) _Float16 v16h;
typedef __attribute__((ext_vector_type(8)))  float    v8f;

constexpr int BN   = 64;    // batch
constexpr int TT   = 12;    // time steps / horizon
constexpr int NN   = 500;   // nodes
constexpr int NP   = 512;   // padded nodes
constexpr int HD   = 64;    // hidden

__device__ __forceinline__ v8f wmma32(v16h a, v16h b, v8f c) {
  return __builtin_amdgcn_wmma_f32_16x16x32_f16(
      false, a, false, b, (short)0, c, false, false);
}

// A-operand k-swizzle within each 32-block: lane-half hf reads pos hf*16..+15
// contiguously and gets k = hf*8+{0..7}, 16+hf*8+{0..7} as WMMA requires.
__device__ __forceinline__ int swz(int k) {
  return (k & ~31) + (k & 7) + ((k & 8) << 1) + ((k & 16) >> 1);
}

// ---------------------------------------------------------------------------
// S = softmax(relu(E E^T), axis=1), f16, padded 512x512, columns k-swizzled.
// ---------------------------------------------------------------------------
__global__ void __launch_bounds__(256) k_supports(const float* __restrict__ E,
                                                  _Float16* __restrict__ S) {
  __shared__ float slog[512];
  __shared__ float red[256];
  const int n = blockIdx.x, tid = threadIdx.x;
  float e[8];
#pragma unroll
  for (int d = 0; d < 8; ++d) e[d] = E[n * 8 + d];
  for (int m = tid; m < NN; m += 256) {
    float s = 0.f;
#pragma unroll
    for (int d = 0; d < 8; ++d) s += e[d] * E[m * 8 + d];
    slog[m] = fmaxf(s, 0.f);
  }
  __syncthreads();
  float lmax = -1e30f;
  for (int m = tid; m < NN; m += 256) lmax = fmaxf(lmax, slog[m]);
  red[tid] = lmax; __syncthreads();
  for (int s = 128; s > 0; s >>= 1) {
    if (tid < s) red[tid] = fmaxf(red[tid], red[tid + s]);
    __syncthreads();
  }
  const float mx = red[0]; __syncthreads();
  float lsum = 0.f;
  for (int m = tid; m < NN; m += 256) lsum += expf(slog[m] - mx);
  red[tid] = lsum; __syncthreads();
  for (int s = 128; s > 0; s >>= 1) {
    if (tid < s) red[tid] += red[tid + s];
    __syncthreads();
  }
  const float inv = 1.f / red[0];
  __syncthreads();
  for (int m = tid; m < NN; m += 256)
    S[(size_t)n * NP + swz(m)] = (_Float16)(expf(slog[m] - mx) * inv);
}

// ---------------------------------------------------------------------------
// Per-node weights, B-fragment-ready: Wt[((n*KB+kb)*O+o)*32 + swz(kloc)], kloc
// k-swizzled to match the A operand.  K zero-padded to Kp.
// ---------------------------------------------------------------------------
__global__ void __launch_bounds__(256) k_make_weights(
    const float* __restrict__ E, const float* __restrict__ Wp,
    const float* __restrict__ bp, _Float16* __restrict__ Wt,
    float* __restrict__ bn, int I, int O, int Kp) {
  const int n = blockIdx.x;
  const int KB = Kp >> 5;
  float e[8];
#pragma unroll
  for (int d = 0; d < 8; ++d) e[d] = E[n * 8 + d];
  const int tot = Kp * O;
  for (int idx = threadIdx.x; idx < tot; idx += blockDim.x) {
    const int o = idx % O, kk = idx / O;
    float v = 0.f;
    if (kk < 2 * I) {
      const int k = (kk < I) ? 0 : 1;
      const int i = kk - k * I;
#pragma unroll
      for (int d = 0; d < 8; ++d)
        v += e[d] * Wp[(((size_t)d * 2 + k) * I + i) * O + o];
    }
    const int kb = kk >> 5, kloc = kk & 31;
    Wt[(((size_t)n * KB + kb) * O + o) * 32 + swz(kloc)] = (_Float16)v;
  }
  for (int o = threadIdx.x; o < O; o += blockDim.x) {
    float v = 0.f;
#pragma unroll
    for (int d = 0; d < 8; ++d) v += e[d] * bp[(size_t)d * O + o];
    bn[(size_t)n * O + o] = v;
  }
}

// ---------------------------------------------------------------------------
// Elementwise helpers.  xg[b][NP][Kp] combined-K (k-swizzled per 32-block);
// xhtc[col][NP] column-major transposed copy for the graph GEMM B-operand.
// ---------------------------------------------------------------------------
__global__ void k_cvt_tw(const float* __restrict__ src, _Float16* __restrict__ dst,
                         int total) {
  int idx = blockIdx.x * blockDim.x + threadIdx.x;
  if (idx < total) dst[idx] = (_Float16)src[(size_t)idx * 2 + 1];
}

__global__ void k_concat_gate(const _Float16* __restrict__ x, int t, int Cin,
                              const float* __restrict__ h,
                              _Float16* __restrict__ xg, _Float16* __restrict__ xhtc,
                              int I, int Kp, int total) {
  int idx = blockIdx.x * blockDim.x + threadIdx.x;
  if (idx >= total) return;
  const int i = idx % I, rem = idx / I, n = rem % NN, b = rem / NN;
  _Float16 v;
  if (i < Cin) v = x[(((size_t)b * TT + t) * NN + n) * Cin + i];
  else         v = (_Float16)h[((size_t)b * NN + n) * HD + (i - Cin)];
  xg[((size_t)b * NP + n) * Kp + swz(i)] = v;
  xhtc[(size_t)(b * I + i) * NP + swz(n)] = v;
}

__global__ void k_concat_upd(const _Float16* __restrict__ x, int t, int Cin,
                             const float* __restrict__ h,
                             const float* __restrict__ zr,
                             _Float16* __restrict__ xg, _Float16* __restrict__ xhtc,
                             int I, int Kp, int total) {
  int idx = blockIdx.x * blockDim.x + threadIdx.x;
  if (idx >= total) return;
  const int i = idx % I, rem = idx / I, n = rem % NN, b = rem / NN;
  _Float16 v;
  if (i < Cin) v = x[(((size_t)b * TT + t) * NN + n) * Cin + i];
  else {
    const int o = i - Cin;
    const float z = zr[((size_t)b * NN + n) * 128 + o];
    v = (_Float16)(z * h[((size_t)b * NN + n) * HD + o]);
  }
  xg[((size_t)b * NP + n) * Kp + swz(i)] = v;
  xhtc[(size_t)(b * I + i) * NP + swz(n)] = v;
}

__global__ void k_update_h(const float* __restrict__ zr, const float* __restrict__ hc,
                           float* __restrict__ h, _Float16* __restrict__ hs,
                           int t, int total) {
  int idx = blockIdx.x * blockDim.x + threadIdx.x;
  if (idx >= total) return;
  const int o = idx % HD, rem = idx / HD, n = rem % NN, b = rem / NN;
  const float r = zr[((size_t)b * NN + n) * 128 + 64 + o];
  const float hn = r * h[idx] + (1.f - r) * hc[idx];
  h[idx] = hn;
  hs[(((size_t)b * TT + t) * NN + n) * HD + o] = (_Float16)hn;
}

// ---------------------------------------------------------------------------
// Graph propagation WMMA GEMM:
//   xg[b][m][swz(I+c)] = sum_k S[m,k] * xhtc[col][k],  col = b*I+c
// Each fragment = one contiguous 32B load.  No bounds checks (padded).
// ---------------------------------------------------------------------------
__global__ void __launch_bounds__(256) k_spmm(const _Float16* __restrict__ S,
                                              const _Float16* __restrict__ xhtc,
                                              _Float16* __restrict__ xg,
                                              int I, int Kp) {
  const int ncols = BN * I;
  const int wave = threadIdx.x >> 5;
  const int l = threadIdx.x & 31, lm = l & 15, hf = l >> 4;
  const int ntile = blockIdx.x;
  const int col0 = (blockIdx.y * 8 + wave) * 16;
  if (col0 >= ncols) return;                 // wave-uniform
  const int col = col0 + lm;
  const int bb = col / I, cc = col % I;
  const int mrow = ntile * 16 + lm;
  const _Float16* arow = S + (size_t)mrow * NP + hf * 16;
  const _Float16* bcol = xhtc + (size_t)col * NP + hf * 16;
  const v8f vz = {0.f, 0.f, 0.f, 0.f, 0.f, 0.f, 0.f, 0.f};
  v8f acc = vz;
  for (int ks = 0; ks < NP; ks += 32) {
    v16h a = *(const v16h*)(arow + ks);
    v16h b = *(const v16h*)(bcol + ks);
    acc = wmma32(a, b, acc);
  }
  _Float16* obase = xg + (size_t)bb * NP * Kp + swz(I + cc);
#pragma unroll
  for (int v = 0; v < 8; ++v) {
    const int m = ntile * 16 + v + 8 * hf;   // pad rows land in unread space
    obase[(size_t)m * Kp] = (_Float16)acc[v];
  }
}

// ---------------------------------------------------------------------------
// Per-node GEMM: out[b,n,o] = act( sum_kk xg[b,n,kk]*W[n,kk,o] + bias[n,o] )
// One block per node; 8 waves tile 64xO.  MODE 0 = sigmoid, 1 = tanh.
// ---------------------------------------------------------------------------
template <int O, int MODE>
__global__ void __launch_bounds__(256) k_gcn_gemm(
    const _Float16* __restrict__ xg, const _Float16* __restrict__ Wt,
    const float* __restrict__ bias, float* __restrict__ out, int Kp) {
  constexpr int ntO = O / 16;
  constexpr int halfO = ntO / 2;             // 4 (O=128) or 2 (O=64)
  const int n = blockIdx.x;
  const int wave = threadIdx.x >> 5;
  const int l = threadIdx.x & 31, lm = l & 15, hf = l >> 4;
  const int mt = wave >> 1;                  // 4 batch tiles (64 rows)
  const int ncb = (wave & 1) * halfO;
  const int bb = mt * 16 + lm;
  const int KB = Kp >> 5;
  const _Float16* xrow = xg + ((size_t)bb * NP + n) * Kp + hf * 16;
  const _Float16* wbase = Wt + (size_t)n * KB * O * 32 + hf * 16;
  const v8f vz = {0.f, 0.f, 0.f, 0.f, 0.f, 0.f, 0.f, 0.f};
  v8f acc[halfO];
#pragma unroll
  for (int j = 0; j < halfO; ++j) acc[j] = vz;

  for (int kb = 0; kb < KB; ++kb) {
    v16h a = *(const v16h*)(xrow + kb * 32);
#pragma unroll
    for (int j = 0; j < halfO; ++j) {
      const int o = (ncb + j) * 16 + lm;
      v16h b = *(const v16h*)(wbase + ((size_t)kb * O + o) * 32);
      acc[j] = wmma32(a, b, acc[j]);
    }
  }
#pragma unroll
  for (int j = 0; j < halfO; ++j) {
    const int o = (ncb + j) * 16 + lm;
    const float bv = bias[(size_t)n * O + o];
#pragma unroll
    for (int v = 0; v < 8; ++v) {
      const int mrow = mt * 16 + v + 8 * hf;
      float r = acc[j][v] + bv;
      r = (MODE == 0) ? (1.f / (1.f + expf(-r))) : tanhf(r);
      out[((size_t)mrow * NN + n) * O + o] = r;
    }
  }
}

// ---------------------------------------------------------------------------
// query[b,:] = out_tw.reshape(B,32000) @ Wq[32000,8]
// ---------------------------------------------------------------------------
__global__ void __launch_bounds__(256) k_query(const float* __restrict__ h,
                                               const float* __restrict__ Wq,
                                               float* __restrict__ q_ws,
                                               float* __restrict__ q_out) {
  __shared__ float red[256 * 8];
  const int b = blockIdx.x, tid = threadIdx.x;
  float acc[8] = {0.f, 0.f, 0.f, 0.f, 0.f, 0.f, 0.f, 0.f};
  for (int idx = tid; idx < NN * HD; idx += 256) {
    const float v = h[(size_t)b * NN * HD + idx];
    const float* w = Wq + (size_t)idx * 8;
#pragma unroll
    for (int d = 0; d < 8; ++d) acc[d] += v * w[d];
  }
#pragma unroll
  for (int d = 0; d < 8; ++d) red[tid * 8 + d] = acc[d];
  __syncthreads();
  for (int s = 128; s > 0; s >>= 1) {
    if (tid < s)
#pragma unroll
      for (int d = 0; d < 8; ++d) red[tid * 8 + d] += red[(tid + s) * 8 + d];
    __syncthreads();
  }
  if (tid < 8) {
    const float v = red[tid];
    q_ws[b * 8 + tid] = v;
    q_out[b * 8 + tid] = v;
  }
}

// ---------------------------------------------------------------------------
// Memory attention: att = softmax(q @ M^T); proto = att @ M; top-2 -> pos/neg
// ---------------------------------------------------------------------------
__global__ void k_attention(const float* __restrict__ q, const float* __restrict__ M,
                            float* __restrict__ proto, float* __restrict__ pos_out,
                            float* __restrict__ neg_out) {
  const int b = threadIdx.x;
  if (b >= BN) return;
  float att[10];
  float mx = -1e30f;
  for (int j = 0; j < 10; ++j) {
    float s = 0.f;
#pragma unroll
    for (int d = 0; d < 8; ++d) s += q[b * 8 + d] * M[j * 8 + d];
    att[j] = s;
    mx = fmaxf(mx, s);
  }
  float sum = 0.f;
  for (int j = 0; j < 10; ++j) { att[j] = expf(att[j] - mx); sum += att[j]; }
  const float inv = 1.f / sum;
  for (int j = 0; j < 10; ++j) att[j] *= inv;
#pragma unroll
  for (int d = 0; d < 8; ++d) {
    float p = 0.f;
    for (int j = 0; j < 10; ++j) p += att[j] * M[j * 8 + d];
    proto[b * 8 + d] = p;
  }
  int i0 = 0; float m0 = -1e30f;
  for (int j = 0; j < 10; ++j) if (att[j] > m0) { m0 = att[j]; i0 = j; }
  int i1 = 0; float m1 = -1e30f;
  for (int j = 0; j < 10; ++j) if (j != i0 && att[j] > m1) { m1 = att[j]; i1 = j; }
#pragma unroll
  for (int d = 0; d < 8; ++d) {
    pos_out[b * 8 + d] = M[i0 * 8 + d];
    neg_out[b * 8 + d] = M[i1 * 8 + d];
  }
}

// Wte1[b,320] = proto @ TE_FC1 ; Wte2[b,5000] = proto @ TE_FC2
__global__ void __launch_bounds__(256) k_wte(const float* __restrict__ proto,
                                             const float* __restrict__ FC1,
                                             const float* __restrict__ FC2,
                                             float* __restrict__ Wte1,
                                             float* __restrict__ Wte2) {
  const int b = blockIdx.x;
  float p[8];
#pragma unroll
  for (int d = 0; d < 8; ++d) p[d] = proto[b * 8 + d];
  for (int j = threadIdx.x; j < 320; j += blockDim.x) {
    float v = 0.f;
#pragma unroll
    for (int d = 0; d < 8; ++d) v += p[d] * FC1[(size_t)d * 320 + j];
    Wte1[(size_t)b * 320 + j] = v;
  }
  for (int j = threadIdx.x; j < 5000; j += blockDim.x) {
    float v = 0.f;
#pragma unroll
    for (int d = 0; d < 8; ++d) v += p[d] * FC2[(size_t)d * 5000 + j];
    Wte2[(size_t)b * 5000 + j] = v;
  }
}

// Xc[b,t,n,0]=source ch0 ; Xc[b,t,n,1]=sum_e tanh(TE@Wte1)[b,t,e]*Wte2[b,e,n]
__global__ void __launch_bounds__(256) k_build_Xc(const float* __restrict__ source,
                                                  const float* __restrict__ TE,
                                                  const float* __restrict__ Wte1,
                                                  const float* __restrict__ Wte2,
                                                  _Float16* __restrict__ Xc) {
  __shared__ float te_sh[10];
  const int blk = blockIdx.x;
  const int b = blk / TT, t = blk % TT;
  const int tid = threadIdx.x;
  if (tid < 10) {
    float s = 0.f;
    for (int i = 0; i < 32; ++i)
      s += TE[((size_t)b * 24 + t) * 32 + i] * Wte1[(size_t)b * 320 + i * 10 + tid];
    te_sh[tid] = tanhf(s);
  }
  __syncthreads();
  for (int n = tid; n < NN; n += blockDim.x) {
    float v = 0.f;
#pragma unroll
    for (int e = 0; e < 10; ++e)
      v += te_sh[e] * Wte2[(size_t)b * 5000 + e * 500 + n];
    const size_t base = (((size_t)b * TT + t) * NN + n) * 2;
    Xc[base + 0] = (_Float16)source[base + 0];
    Xc[base + 1] = (_Float16)v;
  }
}

// y[b,c,n] = tanh(out[b,n,:] . Wc[c,:] + bc[c])
__global__ void __launch_bounds__(256) k_y(const float* __restrict__ h,
                                           const float* __restrict__ Wc,
                                           const float* __restrict__ bc,
                                           float* __restrict__ y) {
  const int blk = blockIdx.x;
  const int b = blk / TT, c = blk % TT;
  const float bias = bc[c];
  for (int n = threadIdx.x; n < NN; n += blockDim.x) {
    float s = bias;
    const float* hr = h + ((size_t)b * NN + n) * HD;
#pragma unroll 8
    for (int k = 0; k < HD; ++k) s += hr[k] * Wc[c * HD + k];
    y[(((size_t)b * TT + c) * NN) + n] = tanhf(s);
  }
}

// ---------------------------------------------------------------------------
// Host side
// ---------------------------------------------------------------------------
namespace {
struct Scratch {
  float* h; _Float16* xg; _Float16* xhtc; float* zr; float* hc;
};

inline dim3 ge(int n) { return dim3((n + 255) / 256); }

void run_layer(hipStream_t stream, const _Float16* x_in, int Cin,
               const _Float16* Wg, const float* bg,
               const _Float16* Wu, const float* bu, int Kp,
               _Float16* hs_out, const _Float16* S, const Scratch& s) {
  const int I = Cin + HD;
  hipMemsetAsync(s.h, 0, sizeof(float) * (size_t)BN * NN * HD, stream);
  hipMemsetAsync(s.xg, 0, (size_t)BN * NP * Kp * 2, stream);          // zero K-pad
  hipMemsetAsync(s.xhtc, 0, (size_t)BN * I * NP * 2, stream);         // zero row-pad
  const int totXH = BN * NN * I;
  const dim3 gXH = ge(totXH);
  const dim3 gH = ge(BN * NN * HD);
  const dim3 gSp(NP / 16, (BN * I + 127) / 128);
  for (int t = 0; t < TT; ++t) {
    k_concat_gate<<<gXH, 256, 0, stream>>>(x_in, t, Cin, s.h, s.xg, s.xhtc, I, Kp, totXH);
    k_spmm<<<gSp, 256, 0, stream>>>(S, s.xhtc, s.xg, I, Kp);
    k_gcn_gemm<128, 0><<<NN, 256, 0, stream>>>(s.xg, Wg, bg, s.zr, Kp);
    k_concat_upd<<<gXH, 256, 0, stream>>>(x_in, t, Cin, s.h, s.zr, s.xg, s.xhtc, I, Kp, totXH);
    k_spmm<<<gSp, 256, 0, stream>>>(S, s.xhtc, s.xg, I, Kp);
    k_gcn_gemm<64, 1><<<NN, 256, 0, stream>>>(s.xg, Wu, bu, s.hc, Kp);
    k_update_h<<<gH, 256, 0, stream>>>(s.zr, s.hc, s.h, hs_out, t, BN * NN * HD);
  }
}
} // namespace

extern "C" void kernel_launch(void* const* d_in, const int* in_sizes, int n_in,
                              void* d_out, int out_size, void* d_ws, size_t ws_size,
                              hipStream_t stream) {
  const float* source = (const float*)d_in[0];
  const float* TE     = (const float*)d_in[1];
  const float* E      = (const float*)d_in[2];
  const float* twgW0  = (const float*)d_in[3];
  const float* twgb0  = (const float*)d_in[4];
  const float* twuW0  = (const float*)d_in[5];
  const float* twub0  = (const float*)d_in[6];
  const float* twgW1  = (const float*)d_in[7];
  const float* twgb1  = (const float*)d_in[8];
  const float* twuW1  = (const float*)d_in[9];
  const float* twub1  = (const float*)d_in[10];
  const float* egW0   = (const float*)d_in[11];
  const float* egb0   = (const float*)d_in[12];
  const float* euW0   = (const float*)d_in[13];
  const float* eub0   = (const float*)d_in[14];
  const float* egW1   = (const float*)d_in[15];
  const float* egb1   = (const float*)d_in[16];
  const float* euW1   = (const float*)d_in[17];
  const float* eub1   = (const float*)d_in[18];
  const float* Memory = (const float*)d_in[19];
  const float* Wq     = (const float*)d_in[20];
  const float* TE_FC1 = (const float*)d_in[21];
  const float* TE_FC2 = (const float*)d_in[22];
  const float* ecW    = (const float*)d_in[23];
  const float* ecb    = (const float*)d_in[24];

  float* y_out   = (float*)d_out;                 // 64*12*500
  float* q_out   = (float*)d_out + 384000;        // 64*8
  float* pos_out = (float*)d_out + 384512;
  float* neg_out = (float*)d_out + 385024;

  // --- workspace carving ---
  char* base = (char*)d_ws;
  size_t off = 0;
  auto take = [&](size_t bytes) -> void* {
    void* p = base + off;
    off = (off + bytes + 255) & ~(size_t)255;
    return p;
  };
  _Float16* S     = (_Float16*)take((size_t)NP * NP * 2);
  _Float16* tw_in = (_Float16*)take((size_t)BN * TT * NN * 2);
  _Float16* Xc    = (_Float16*)take((size_t)BN * TT * NN * 2 * 2);
  _Float16* hsA   = (_Float16*)take((size_t)BN * TT * NN * HD * 2);
  _Float16* hsB   = (_Float16*)take((size_t)BN * TT * NN * HD * 2);
  Scratch sc;
  sc.h    = (float*)take((size_t)BN * NN * HD * 4);
  sc.xg   = (_Float16*)take((size_t)BN * NP * 256 * 2);
  sc.xhtc = (_Float16*)take((size_t)BN * 128 * NP * 2);
  sc.zr   = (float*)take((size_t)BN * NN * 128 * 4);
  sc.hc   = (float*)take((size_t)BN * NN * HD * 4);

  auto takeW = [&](int Kp, int O) { return (_Float16*)take((size_t)NN * Kp * O * 2); };
  auto takeB = [&](int O) { return (float*)take((size_t)NN * O * 4); };
  _Float16* Wg_tw0 = takeW(160, 128); float* bg_tw0 = takeB(128);
  _Float16* Wu_tw0 = takeW(160, 64);  float* bu_tw0 = takeB(64);
  _Float16* Wg_tw1 = takeW(256, 128); float* bg_tw1 = takeB(128);
  _Float16* Wu_tw1 = takeW(256, 64);  float* bu_tw1 = takeB(64);
  _Float16* Wg_e0 = takeW(160, 128); float* bg_e0 = takeB(128);
  _Float16* Wu_e0 = takeW(160, 64);  float* bu_e0 = takeB(64);
  _Float16* Wg_e1 = takeW(256, 128); float* bg_e1 = takeB(128);
  _Float16* Wu_e1 = takeW(256, 64);  float* bu_e1 = takeB(64);
  float* q_ws  = (float*)take((size_t)BN * 8 * 4);
  float* proto = (float*)take((size_t)BN * 8 * 4);
  float* Wte1  = (float*)take((size_t)BN * 320 * 4);
  float* Wte2  = (float*)take((size_t)BN * 5000 * 4);
  (void)ws_size; (void)n_in; (void)in_sizes; (void)out_size;

  // --- supports (padded, swizzled) + per-node fragment-ready weights ---
  hipMemsetAsync(S, 0, (size_t)NP * NP * 2, stream);
  k_supports<<<NN, 256, 0, stream>>>(E, S);
  k_make_weights<<<NN, 256, 0, stream>>>(E, twgW0, twgb0, Wg_tw0, bg_tw0, 65, 128, 160);
  k_make_weights<<<NN, 256, 0, stream>>>(E, twuW0, twub0, Wu_tw0, bu_tw0, 65, 64, 160);
  k_make_weights<<<NN, 256, 0, stream>>>(E, twgW1, twgb1, Wg_tw1, bg_tw1, 128, 128, 256);
  k_make_weights<<<NN, 256, 0, stream>>>(E, twuW1, twub1, Wu_tw1, bu_tw1, 128, 64, 256);
  k_make_weights<<<NN, 256, 0, stream>>>(E, egW0, egb0, Wg_e0, bg_e0, 66, 128, 160);
  k_make_weights<<<NN, 256, 0, stream>>>(E, euW0, eub0, Wu_e0, bu_e0, 66, 64, 160);
  k_make_weights<<<NN, 256, 0, stream>>>(E, egW1, egb1, Wg_e1, bg_e1, 128, 128, 256);
  k_make_weights<<<NN, 256, 0, stream>>>(E, euW1, eub1, Wu_e1, bu_e1, 128, 64, 256);

  // --- time-of-week branch ---
  k_cvt_tw<<<ge(BN * TT * NN), 256, 0, stream>>>(source, tw_in, BN * TT * NN);
  run_layer(stream, tw_in, 1, Wg_tw0, bg_tw0, Wu_tw0, bu_tw0, 160, hsA, S, sc);
  run_layer(stream, hsA, 64, Wg_tw1, bg_tw1, Wu_tw1, bu_tw1, 256, hsB, S, sc);

  // --- memory attention ---
  k_query<<<BN, 256, 0, stream>>>(sc.h, Wq, q_ws, q_out);
  k_attention<<<1, 64, 0, stream>>>(q_ws, Memory, proto, pos_out, neg_out);
  k_wte<<<BN, 256, 0, stream>>>(proto, TE_FC1, TE_FC2, Wte1, Wte2);
  k_build_Xc<<<BN * TT, 256, 0, stream>>>(source, TE, Wte1, Wte2, Xc);

  // --- encoder branch ---
  run_layer(stream, Xc, 2, Wg_e0, bg_e0, Wu_e0, bu_e0, 160, hsA, S, sc);
  run_layer(stream, hsA, 64, Wg_e1, bg_e1, Wu_e1, bu_e1, 256, hsB, S, sc);

  // --- output projection ---
  k_y<<<BN * TT, 256, 0, stream>>>(sc.h, ecW, ecb, y_out);
}
